// MARN1_newz_20504173871331
// MI455X (gfx1250) — compile-verified, matrix operand order checked by reading, not run on hardware
//
#include <hip/hip_runtime.h>
#include <hip/hip_bf16.h>
#include <math.h>

// ---------------------------------------------------------------------------
// MARN1 forward for MI455X (gfx1250, wave32, WMMA f16 with fp32 accumulate).
//
// Input index map (setup_inputs dict order, leaves in insertion order):
//   0: x (128,256,200)   1: qmask (128,256,2)
//   enc_l @2..14 : wq wk wv wo bo ln1_g ln1_b w1 b1 w2 b2 ln2_g ln2_b
//   enc_a @15..27: same
//   lsthm_l @28..33: W bW U bU V bV       lsthm_a @34..39
//   40: w(2) 41: S_w 42: S_b 43: att_w 44: att_b 45: attl_w 46: attl_b
//   47: atta_w 48: atta_b 49: cm_w 50: cm_b 51: fcz_w 52: fcz_b
//   53: o1_w 54: o1_b 55: o2_w 56: o2_b
//
// d_out = [output (256*128,6) | xa_t (128,256,100) | xl_t (128,256,100)]
//
// Strategy: weights are reused 128..2048x per call and total ~3MB (L2
// resident at 192MB), so we pre-transpose+convert every weight matrix to f16
// (N x Kp row-major, Kp = K rounded to 8, zero padded). The GEMM inner loop
// is then: 4x float4 A loads (+cvt) and 2x 8-half b128 B loads per
// v_wmma_f32_16x16x32_f16 — no strided gathers, no guarded loads. Edge tiles
// use index-clamped loads with value selects (no EXEC divergence).
// ---------------------------------------------------------------------------

typedef __attribute__((ext_vector_type(16))) _Float16 v16h;
typedef __attribute__((ext_vector_type(8)))  _Float16 v8h;
typedef __attribute__((ext_vector_type(8)))  float    v8f;

__device__ __forceinline__ v8f wmma_f16(v16h a, v16h b, v8f c) {
  return __builtin_amdgcn_wmma_f32_16x16x32_f16(false, a, false, b, (short)0, c,
                                                false, false);
}
__device__ __forceinline__ float sigf(float x) { return 1.0f / (1.0f + expf(-x)); }

// ---------------------------------------------------------------------------
// GEMM: C = act(alpha*(A @ B) + bias + addend)
// A: MxK fp32 (lda). Bh: f16, transposed+padded (N rows of Kp).
// One wave per 16x16 tile. Fragment layout per CDNA5 ISA 7.12.2:
// lane = M (A) / N (B) row, halves 0..7 -> K kb..kb+7, 8..15 -> K+16 run,
// kb = 8 for lanes 16..31.
// ---------------------------------------------------------------------------
__global__ void __launch_bounds__(32) gemm_kernel(
    const float* __restrict__ A, int lda,
    const _Float16* __restrict__ Bh, int Kp,
    const float* __restrict__ bias,
    const float* __restrict__ addend, int ldadd,
    float* __restrict__ C, int ldc,
    int M, int K, int N, float alpha, int act)
{
  int lane = threadIdx.x;
  int n0 = blockIdx.x * 16, m0 = blockIdx.y * 16;
  int r  = lane & 15;
  int kb = (lane >> 4) * 8;
  int arow = m0 + r, bcol = n0 + r;

  bool full  = (m0 + 16 <= M) && (n0 + 16 <= N);   // block-uniform
  int  Kfull = full ? (K & ~31) : 0;

  const float*    Arow = A  + (size_t)arow * lda;
  const _Float16* Brow = Bh + (size_t)bcol * Kp;

  v8f acc = {};

  // fast path: vectorized, unguarded
  for (int kc = 0; kc < Kfull; kc += 32) {
    __builtin_prefetch(Arow + kc + 160, 0, 1);          // global_prefetch_b8
    float4 a0 = *(const float4*)(Arow + kc + kb);
    float4 a1 = *(const float4*)(Arow + kc + kb + 4);
    float4 a2 = *(const float4*)(Arow + kc + kb + 16);
    float4 a3 = *(const float4*)(Arow + kc + kb + 20);
    v8h b0 = *(const v8h*)(Brow + kc + kb);
    v8h b1 = *(const v8h*)(Brow + kc + kb + 16);
    v16h a, b;
    a[0] = (_Float16)a0.x; a[1] = (_Float16)a0.y; a[2] = (_Float16)a0.z; a[3] = (_Float16)a0.w;
    a[4] = (_Float16)a1.x; a[5] = (_Float16)a1.y; a[6] = (_Float16)a1.z; a[7] = (_Float16)a1.w;
    a[8] = (_Float16)a2.x; a[9] = (_Float16)a2.y; a[10] = (_Float16)a2.z; a[11] = (_Float16)a2.w;
    a[12] = (_Float16)a3.x; a[13] = (_Float16)a3.y; a[14] = (_Float16)a3.z; a[15] = (_Float16)a3.w;
#pragma unroll
    for (int i = 0; i < 8; ++i) { b[i] = b0[i]; b[8 + i] = b1[i]; }
    acc = wmma_f16(a, b, acc);
  }

  // edge/tail path: clamped unconditional loads + value selects (no branches)
  if (Kfull < K) {
    bool aok = arow < M, bok = bcol < N;
    const float*    Ar = A  + (size_t)(aok ? arow : (M - 1)) * lda;
    const _Float16* Br = Bh + (size_t)(bok ? bcol : (N - 1)) * Kp;
    for (int kc = Kfull; kc < K; kc += 32) {
      v16h a, b;
#pragma unroll
      for (int h = 0; h < 16; ++h) {
        int k  = kc + kb + (h < 8 ? h : (8 + h));
        int ka = k < K  ? k : (K - 1);
        int kbn = k < Kp ? k : (Kp - 1);
        float    av = Ar[ka];
        _Float16 bv = Br[kbn];
        bool ok = k < K;
        a[h] = (aok && ok) ? (_Float16)av : (_Float16)0.f;
        b[h] = (bok && ok) ? bv : (_Float16)0.f;
      }
      acc = wmma_f16(a, b, acc);
    }
  }

  int rhi = (lane >> 4) * 8;
  int col = n0 + (lane & 15);
#pragma unroll
  for (int v = 0; v < 8; ++v) {
    int row = m0 + v + rhi;
    if (row < M && col < N) {
      float xv = acc[v] * alpha;
      if (bias)   xv += bias[col];
      if (addend) xv += addend[(size_t)row * ldadd + col];
      if (act == 1) xv = fmaxf(xv, 0.0f);
      C[(size_t)row * ldc + col] = xv;
    }
  }
}

// ---------------------------------------------------------------------------
// Fused MHA for one (sequence, head). T=128, DK=DV=40, row stride 320.
// Entirely in LDS (~110 KB of the 320 KB/WGP). Q/K padded to 64 halves so
// all fragment reads are contiguous b128; V staged transposed (VT[48][128]).
// ---------------------------------------------------------------------------
__global__ void __launch_bounds__(256) attn_kernel(
    const float* __restrict__ QB, const float* __restrict__ KB,
    const float* __restrict__ VB, float* __restrict__ AO)
{
  const int h = blockIdx.x & 7;
  const int s = blockIdx.x >> 3;
  __shared__ _Float16 qs[128][64];
  __shared__ _Float16 ks[128][64];
  __shared__ _Float16 vt[48][128];
  __shared__ float    sS[128][132];

  int tid = threadIdx.x;
  for (int idx = tid; idx < 128 * 40; idx += 256) {
    int rr = idx / 40, cc = idx % 40;
    size_t base = ((size_t)s * 128 + rr) * 320 + h * 40 + cc;
    qs[rr][cc] = (_Float16)QB[base];
    ks[rr][cc] = (_Float16)KB[base];
    vt[cc][rr] = (_Float16)VB[base];
  }
  for (int idx = tid; idx < 128 * 24; idx += 256) {
    int rr = idx / 24, cc = 40 + idx % 24;
    qs[rr][cc] = (_Float16)0.f;
    ks[rr][cc] = (_Float16)0.f;
  }
  for (int idx = tid; idx < 8 * 128; idx += 256)
    vt[40 + idx / 128][idx % 128] = (_Float16)0.f;
  __syncthreads();

  int wave = tid >> 5, lane = tid & 31;
  int r = lane & 15, kb = (lane >> 4) * 8, rhi = (lane >> 4) * 8;

  // S = (Q @ K^T) / sqrt(40):  8x8 tiles over 8 waves
  for (int tile = wave; tile < 64; tile += 8) {
    int ti = tile >> 3, tj = tile & 7;
    v8f acc = {};
#pragma unroll
    for (int kc = 0; kc < 64; kc += 32) {
      v8h a0 = *(const v8h*)&qs[ti * 16 + r][kc + kb];
      v8h a1 = *(const v8h*)&qs[ti * 16 + r][kc + kb + 16];
      v8h b0 = *(const v8h*)&ks[tj * 16 + r][kc + kb];
      v8h b1 = *(const v8h*)&ks[tj * 16 + r][kc + kb + 16];
      v16h a, b;
#pragma unroll
      for (int i = 0; i < 8; ++i) { a[i] = a0[i]; a[8 + i] = a1[i]; b[i] = b0[i]; b[8 + i] = b1[i]; }
      acc = wmma_f16(a, b, acc);
    }
#pragma unroll
    for (int v = 0; v < 8; ++v)
      sS[ti * 16 + v + rhi][tj * 16 + (lane & 15)] = acc[v] * 0.15811388300841898f;
  }
  __syncthreads();

  if (tid < 128) {  // row softmax, vectorized LDS access
    float4* rowp = (float4*)&sS[tid][0];
    float mx = -1e30f;
    for (int j = 0; j < 32; ++j) {
      float4 v = rowp[j];
      mx = fmaxf(mx, fmaxf(fmaxf(v.x, v.y), fmaxf(v.z, v.w)));
    }
    float sum = 0.f;
    for (int j = 0; j < 32; ++j) {
      float4 v = rowp[j];
      v.x = expf(v.x - mx); v.y = expf(v.y - mx);
      v.z = expf(v.z - mx); v.w = expf(v.w - mx);
      sum += v.x + v.y + v.z + v.w;
      rowp[j] = v;
    }
    float inv = 1.0f / sum;
    for (int j = 0; j < 32; ++j) {
      float4 v = rowp[j];
      v.x *= inv; v.y *= inv; v.z *= inv; v.w *= inv;
      rowp[j] = v;
    }
  }
  __syncthreads();

  // O = S @ V : 8 x 3 tiles (cols 40..47 zero pad)
  for (int tile = wave; tile < 24; tile += 8) {
    int ti = tile / 3, tj = tile % 3;
    int col = tj * 16 + r;
    v8f acc = {};
#pragma unroll
    for (int kc = 0; kc < 128; kc += 32) {
      float4 s0 = *(const float4*)&sS[ti * 16 + r][kc + kb];
      float4 s1 = *(const float4*)&sS[ti * 16 + r][kc + kb + 4];
      float4 s2 = *(const float4*)&sS[ti * 16 + r][kc + kb + 16];
      float4 s3 = *(const float4*)&sS[ti * 16 + r][kc + kb + 20];
      v8h b0 = *(const v8h*)&vt[col][kc + kb];
      v8h b1 = *(const v8h*)&vt[col][kc + kb + 16];
      v16h a, b;
      a[0] = (_Float16)s0.x; a[1] = (_Float16)s0.y; a[2] = (_Float16)s0.z; a[3] = (_Float16)s0.w;
      a[4] = (_Float16)s1.x; a[5] = (_Float16)s1.y; a[6] = (_Float16)s1.z; a[7] = (_Float16)s1.w;
      a[8] = (_Float16)s2.x; a[9] = (_Float16)s2.y; a[10] = (_Float16)s2.z; a[11] = (_Float16)s2.w;
      a[12] = (_Float16)s3.x; a[13] = (_Float16)s3.y; a[14] = (_Float16)s3.z; a[15] = (_Float16)s3.w;
#pragma unroll
      for (int i = 0; i < 8; ++i) { b[i] = b0[i]; b[8 + i] = b1[i]; }
      acc = wmma_f16(a, b, acc);
    }
    int ocol = tj * 16 + (lane & 15);
#pragma unroll
    for (int v = 0; v < 8; ++v) {
      int row = ti * 16 + v + rhi;
      if (ocol < 40)
        AO[((size_t)s * 128 + row) * 320 + h * 40 + ocol] = acc[v];
    }
  }
}

// transpose + convert weight: W (KxN fp32 row-major) -> Bh (N x Kp f16, pad 0)
__global__ void wtrans_kernel(const float* __restrict__ W, _Float16* __restrict__ Bh,
                              int K, int N, int Kp)
{
  int gid = blockIdx.x * blockDim.x + threadIdx.x;
  if (gid >= N * Kp) return;
  int n = gid / Kp, k = gid % Kp;
  Bh[gid] = (k < K) ? (_Float16)W[(size_t)k * N + n] : (_Float16)0.f;
}

// LayerNorm over D; out row address = base + (row/128)*os + (row%128)*ot.
__global__ void __launch_bounds__(128) ln_kernel(
    const float* __restrict__ X, const float* __restrict__ g,
    const float* __restrict__ b, float* __restrict__ out,
    int D, int os, int ot)
{
  int row = blockIdx.x, tid = threadIdx.x;
  __shared__ float red[128];
  const float* xr = X + (size_t)row * D;
  float sum = 0.f;
  for (int d = tid; d < D; d += 128) sum += xr[d];
  red[tid] = sum; __syncthreads();
  for (int s = 64; s > 0; s >>= 1) { if (tid < s) red[tid] += red[tid + s]; __syncthreads(); }
  float mean = red[0] / D; __syncthreads();
  float vsum = 0.f;
  for (int d = tid; d < D; d += 128) { float t = xr[d] - mean; vsum += t * t; }
  red[tid] = vsum; __syncthreads();
  for (int s = 64; s > 0; s >>= 1) { if (tid < s) red[tid] += red[tid + s]; __syncthreads(); }
  float inv = rsqrtf(red[0] / D + 1e-5f);
  float* orow = out + (size_t)(row / 128) * os + (size_t)(row % 128) * ot;
  for (int d = tid; d < D; d += 128) orow[d] = (xr[d] - mean) * inv * g[d] + b[d];
}

// gather x[:, s0:s0+ns, off:off+100] -> batch-first (ns*128, 100)
__global__ void enc_gather(const float* __restrict__ x, float* __restrict__ Xin,
                           int s0, int ns, int off)
{
  int gid = blockIdx.x * blockDim.x + threadIdx.x;
  int total = ns * 128 * 100;
  if (gid >= total) return;
  int d = gid % 100, i = gid / 100;
  int t = i % 128, sl = i / 128;
  Xin[gid] = x[((size_t)t * 256 + s0 + sl) * 200 + off + d];
}

// pack [W;U;V] per cell -> transposed f16 (512 x 488), and bW+bU+bV (fp32)
__global__ void pack_kernel(
    const float* Wl, const float* bWl, const float* Ul, const float* bUl,
    const float* Vl, const float* bVl,
    const float* Wa, const float* bWa, const float* Ua, const float* bUa,
    const float* Va, const float* bVa,
    _Float16* WUVlT, _Float16* WUVaT, float* bl, float* ba)
{
  const int KP = 488;
  int gid = blockIdx.x * blockDim.x + threadIdx.x;
  const int nw = 512 * KP;
  if (gid < 2 * nw) {
    int cell = gid / nw, i = gid % nw;
    int j = i / KP, k = i % KP;
    const float* W = cell ? Wa : Wl;
    const float* U = cell ? Ua : Ul;
    const float* V = cell ? Va : Vl;
    float v = 0.f;
    if (k < 100)      v = W[k * 512 + j];
    else if (k < 228) v = U[(k - 100) * 512 + j];
    else if (k < 484) v = V[(k - 228) * 512 + j];
    (cell ? WUVaT : WUVlT)[i] = (_Float16)v;
  } else {
    int rr = gid - 2 * nw;
    if (rr < 1024) {
      int cell = rr / 512, j = rr % 512;
      (cell ? ba : bl)[j] = cell ? (bWa[j] + bUa[j] + bVa[j])
                                 : (bWl[j] + bUl[j] + bVl[j]);
    }
  }
}

// ------------------------- scan elementwise kernels ------------------------
__global__ void qprep_kernel(const float* __restrict__ qmask, const float* __restrict__ q,
                             float* __restrict__ q0, int t)
{
  int gid = blockIdx.x * blockDim.x + threadIdx.x;
  if (gid >= 256 * 512) return;
  int n = gid >> 9, d = gid & 511;
  const float* qm = qmask + ((size_t)t * 256 + n) * 2;
  int idx = (qm[1] > qm[0]) ? 1 : 0;
  q0[gid] = q[((size_t)n * 2 + idx) * 512 + d];
}

__global__ void qupdate_kernel(const float* __restrict__ qmask, const float* __restrict__ q0,
                               const float* __restrict__ qaff, float* __restrict__ q, int t)
{
  int gid = blockIdx.x * blockDim.x + threadIdx.x;
  if (gid >= 256 * 2 * 512) return;
  int n = gid >> 10, p = (gid >> 9) & 1, d = gid & 511;
  float m = qmask[((size_t)t * 256 + n) * 2 + p];
  q[gid] = q0[n * 512 + d] * (1.0f - m) + qaff[n * 512 + d] * m;
}

__global__ void xcat_kernel(const float* __restrict__ xl_t, const float* __restrict__ xa_t,
                            const float* __restrict__ allhs,
                            float* __restrict__ XcatL, float* __restrict__ XcatA, int t)
{
  int gid = blockIdx.x * blockDim.x + threadIdx.x;
  const int per = 256 * 484;
  if (gid >= 2 * per) return;
  int half = gid / per, i = gid % per;
  int n = i / 484, k = i % 484;
  float v;
  if (k < 100)      v = (half ? xa_t : xl_t)[(size_t)t * 25600 + n * 100 + k];
  else if (k < 228) v = allhs[(size_t)n * 512 + (half ? 128 : 0) + (k - 100)];
  else              v = allhs[(size_t)n * 512 + 256 + (k - 228)];
  (half ? XcatA : XcatL)[i] = v;
}

__global__ void lsthm_kernel(const float* __restrict__ gatesL, const float* __restrict__ gatesA,
                             float* __restrict__ c_cat, float* __restrict__ allhs)
{
  int gid = blockIdx.x * blockDim.x + threadIdx.x;
  if (gid >= 256 * 256) return;
  int n = gid >> 8, col = gid & 255;
  int cell = col >> 7, j = col & 127;
  const float* g = (cell ? gatesA : gatesL) + (size_t)n * 512;
  float f = g[j], ii = g[128 + j], o = g[256 + j], ch = g[384 + j];
  float c2 = sigf(f) * c_cat[gid] + sigf(ii) * tanhf(ch);
  c_cat[gid] = c2;
  allhs[(size_t)n * 512 + col] = tanhf(c2) * sigf(o);
}

__global__ void __launch_bounds__(128) attend_kernel(
    const float* __restrict__ attraw, const float* __restrict__ attlraw,
    const float* __restrict__ attaraw, const float* __restrict__ c_cat,
    float* __restrict__ crossin, float* __restrict__ zflat)
{
  int a = blockIdx.x >> 8, n = blockIdx.x & 255;
  int tid = threadIdx.x;
  __shared__ float red[128];

  // att over cs (256 wide)
  float v0 = attraw[(size_t)n * 1024 + a * 256 + tid];
  float v1 = attraw[(size_t)n * 1024 + a * 256 + 128 + tid];
  red[tid] = fmaxf(v0, v1); __syncthreads();
  for (int s = 64; s > 0; s >>= 1) { if (tid < s) red[tid] = fmaxf(red[tid], red[tid + s]); __syncthreads(); }
  float mx = red[0]; __syncthreads();
  float e0 = expf(v0 - mx), e1 = expf(v1 - mx);
  red[tid] = e0 + e1; __syncthreads();
  for (int s = 64; s > 0; s >>= 1) { if (tid < s) red[tid] += red[tid + s]; __syncthreads(); }
  float inv = 1.0f / red[0]; __syncthreads();
  zflat[(size_t)n * 1600 + a * 400 + 144 + tid]        = e0 * inv * c_cat[(size_t)n * 256 + tid];
  zflat[(size_t)n * 1600 + a * 400 + 144 + 128 + tid]  = e1 * inv * c_cat[(size_t)n * 256 + 128 + tid];

  // att_l relu-softmax (128)
  float l = fmaxf(attlraw[(size_t)n * 512 + a * 128 + tid], 0.0f);
  red[tid] = l; __syncthreads();
  for (int s = 64; s > 0; s >>= 1) { if (tid < s) red[tid] = fmaxf(red[tid], red[tid + s]); __syncthreads(); }
  mx = red[0]; __syncthreads();
  float el = expf(l - mx);
  red[tid] = el; __syncthreads();
  for (int s = 64; s > 0; s >>= 1) { if (tid < s) red[tid] += red[tid + s]; __syncthreads(); }
  inv = 1.0f / red[0]; __syncthreads();
  crossin[((size_t)a * 256 + n) * 256 + tid] = el * inv * c_cat[(size_t)n * 256 + tid];

  // att_a relu-softmax (128)
  float aa = fmaxf(attaraw[(size_t)n * 512 + a * 128 + tid], 0.0f);
  red[tid] = aa; __syncthreads();
  for (int s = 64; s > 0; s >>= 1) { if (tid < s) red[tid] = fmaxf(red[tid], red[tid + s]); __syncthreads(); }
  mx = red[0]; __syncthreads();
  float ea = expf(aa - mx);
  red[tid] = ea; __syncthreads();
  for (int s = 64; s > 0; s >>= 1) { if (tid < s) red[tid] += red[tid + s]; __syncthreads(); }
  inv = 1.0f / red[0];
  crossin[((size_t)a * 256 + n) * 256 + 128 + tid] = ea * inv * c_cat[(size_t)n * 256 + 128 + tid];
}

__global__ void zfill_kernel(const float* __restrict__ crossbuf, float* __restrict__ zflat)
{
  int gid = blockIdx.x * blockDim.x + threadIdx.x;
  if (gid >= 1024 * 144) return;
  int rr = gid / 144, j = gid % 144;
  int a = rr >> 8, n = rr & 255;
  zflat[(size_t)n * 1600 + a * 400 + j] = crossbuf[gid];
}

__global__ void znew_kernel(const float* __restrict__ wsm, const float* __restrict__ zrelu,
                            float* __restrict__ allhs)
{
  int gid = blockIdx.x * blockDim.x + threadIdx.x;
  if (gid >= 256 * 256) return;
  int n = gid >> 8, j = gid & 255;
  float a = wsm[0], b = wsm[1];
  float m = fmaxf(a, b);
  float e1 = expf(a - m), e2 = expf(b - m);
  float inv = 1.0f / (e1 + e2);
  float* zp = allhs + (size_t)n * 512 + 256 + j;
  *zp = (e1 * inv) * zrelu[gid] + (e2 * inv) * (*zp);
}

// ---------------------------------------------------------------------------
extern "C" void kernel_launch(void* const* d_in, const int* in_sizes, int n_in,
                              void* d_out, int out_size, void* d_ws, size_t ws_size,
                              hipStream_t stream)
{
  (void)in_sizes; (void)n_in; (void)out_size;
  auto IN = [&](int i) { return (const float*)d_in[i]; };

  const float* x     = IN(0);
  const float* qmask = IN(1);
  float* out    = (float*)d_out;
  float* xa_out = out + 196608;                 // (128,256,100)
  float* xl_out = out + 196608 + 3276800;       // (128,256,100)

  char* wsb = (char*)d_ws;
  size_t cur = 0;
  auto allocF = [&](size_t n) -> float* {
    cur = (cur + 255) & ~(size_t)255; float* p = (float*)(wsb + cur); cur += n * 4; return p;
  };
  auto allocH = [&](size_t n) -> _Float16* {
    cur = (cur + 255) & ~(size_t)255; _Float16* p = (_Float16*)(wsb + cur); cur += n * 2; return p;
  };
  auto kpad = [](int K) { return (K + 7) & ~7; };

  // --- f16 transposed weights (persist all call) ---
  _Float16* encT[2][6];
  for (int e = 0; e < 2; ++e) {
    encT[e][0] = allocH((size_t)320 * 104);   // wq
    encT[e][1] = allocH((size_t)320 * 104);   // wk
    encT[e][2] = allocH((size_t)320 * 104);   // wv
    encT[e][3] = allocH((size_t)100 * 320);   // wo
    encT[e][4] = allocH((size_t)50  * 104);   // w1
    encT[e][5] = allocH((size_t)100 * 56);    // w2
  }
  _Float16* S_wT   = allocH((size_t)512 * 512);
  _Float16* att_wT = allocH((size_t)1024 * 256);
  _Float16* attl_wT = allocH((size_t)512 * 128);
  _Float16* atta_wT = allocH((size_t)512 * 128);
  _Float16* cm_wT  = allocH((size_t)144 * 256);
  _Float16* fcz_wT = allocH((size_t)256 * 1600);
  _Float16* o1_wT  = allocH((size_t)32 * 512);
  _Float16* o2_wT  = allocH((size_t)6 * 32);
  _Float16* WUVlT  = allocH((size_t)512 * 488);
  _Float16* WUVaT  = allocH((size_t)512 * 488);

  // --- persistent scan scratch (fp32) ---
  float* bWUVl = allocF(512);
  float* bWUVa = allocF(512);
  float* allhs = allocF(256 * 512);   // [h_l | h_a | z]
  float* c_cat = allocF(256 * 256);   // [c_l | c_a]
  float* qst   = allocF(256 * 2 * 512);
  float* q0    = allocF(256 * 512);
  float* qaff  = allocF(256 * 512);
  float* XcatL = allocF(256 * 484);
  float* XcatA = allocF(256 * 484);
  float* gatesL = allocF(256 * 512);
  float* gatesA = allocF(256 * 512);
  float* attraw = allocF(256 * 1024);
  float* attlraw = allocF(256 * 512);
  float* attaraw = allocF(256 * 512);
  float* crossin = allocF(1024 * 256);
  float* crossbuf = allocF(1024 * 144);
  float* zflat = allocF(256 * 1600);
  float* zrelu = allocF(256 * 256);
  float* hid   = allocF(256 * 32);

  // --- encoder scratch: chunk sequences by remaining workspace ---
  size_t remainF = (ws_size > cur) ? (ws_size - cur - 4096) / 4 : 0;
  const size_t perseq = 1730ull * 128ull;
  int cands[7] = {64, 32, 16, 8, 4, 2, 1};
  int ns = 1;
  for (int i = 0; i < 7; ++i) { if ((size_t)cands[i] * perseq <= remainF) { ns = cands[i]; break; } }
  int Mtok = ns * 128;
  float* Xin = allocF((size_t)Mtok * 100);
  float* QB  = allocF((size_t)Mtok * 320);
  float* KB  = allocF((size_t)Mtok * 320);
  float* VB  = allocF((size_t)Mtok * 320);
  float* AO  = allocF((size_t)Mtok * 320);
  float* O1  = allocF((size_t)Mtok * 100);
  float* ON  = allocF((size_t)Mtok * 100);
  float* F1  = allocF((size_t)Mtok * 50);
  float* F2  = allocF((size_t)Mtok * 100);

  auto GEMM = [&](const float* A, int lda, const _Float16* Bh, int Kp,
                  const float* bias, const float* add, int ldadd,
                  float* C, int ldc, int M, int K, int N, float alpha, int act) {
    dim3 g((N + 15) / 16, (M + 15) / 16);
    gemm_kernel<<<g, dim3(32), 0, stream>>>(A, lda, Bh, Kp, bias, add, ldadd,
                                            C, ldc, M, K, N, alpha, act);
  };
  auto WT = [&](const float* W, int K, int N, _Float16* dst) {
    int Kp = kpad(K); int tot = N * Kp;
    wtrans_kernel<<<(tot + 255) / 256, 256, 0, stream>>>(W, dst, K, N, Kp);
  };

  // init state, pack + transpose all weights once
  hipMemsetAsync(allhs, 0, 256 * 512 * sizeof(float), stream);
  hipMemsetAsync(c_cat, 0, 256 * 256 * sizeof(float), stream);
  hipMemsetAsync(qst,   0, 256 * 2 * 512 * sizeof(float), stream);
  {
    int tot = 2 * 512 * 488 + 1024;
    pack_kernel<<<(tot + 255) / 256, 256, 0, stream>>>(
        IN(28), IN(29), IN(30), IN(31), IN(32), IN(33),
        IN(34), IN(35), IN(36), IN(37), IN(38), IN(39),
        WUVlT, WUVaT, bWUVl, bWUVa);
  }
  for (int e = 0; e < 2; ++e) {
    int pb = e ? 15 : 2;
    WT(IN(pb + 0), 100, 320, encT[e][0]);
    WT(IN(pb + 1), 100, 320, encT[e][1]);
    WT(IN(pb + 2), 100, 320, encT[e][2]);
    WT(IN(pb + 3), 320, 100, encT[e][3]);
    WT(IN(pb + 7), 100, 50,  encT[e][4]);
    WT(IN(pb + 9), 50,  100, encT[e][5]);
  }
  WT(IN(41), 512, 512, S_wT);
  WT(IN(43), 256, 1024, att_wT);
  WT(IN(45), 128, 512, attl_wT);
  WT(IN(47), 128, 512, atta_wT);
  WT(IN(49), 256, 144, cm_wT);
  WT(IN(51), 1600, 256, fcz_wT);
  WT(IN(53), 512, 32, o1_wT);
  WT(IN(55), 32, 6, o2_wT);

  // -------------------- encoders --------------------
  auto run_encoder = [&](int e, int off, float* xout) {
    int pb = e ? 15 : 2;
    const float* bo  = IN(pb + 4);
    const float* g1  = IN(pb + 5); const float* b1n = IN(pb + 6);
    const float* bb1 = IN(pb + 8); const float* bb2 = IN(pb + 10);
    const float* g2  = IN(pb + 11); const float* b2n = IN(pb + 12);
    for (int s0 = 0; s0 < 256; s0 += ns) {
      int tot = Mtok * 100;
      enc_gather<<<(tot + 255) / 256, 256, 0, stream>>>(x, Xin, s0, ns, off);
      GEMM(Xin, 100, encT[e][0], 104, nullptr, nullptr, 0, QB, 320, Mtok, 100, 320, 1.f, 0);
      GEMM(Xin, 100, encT[e][1], 104, nullptr, nullptr, 0, KB, 320, Mtok, 100, 320, 1.f, 0);
      GEMM(Xin, 100, encT[e][2], 104, nullptr, nullptr, 0, VB, 320, Mtok, 100, 320, 1.f, 0);
      attn_kernel<<<ns * 8, 256, 0, stream>>>(QB, KB, VB, AO);
      GEMM(AO, 320, encT[e][3], 320, bo, Xin, 100, O1, 100, Mtok, 320, 100, 1.f, 0);
      ln_kernel<<<Mtok, 128, 0, stream>>>(O1, g1, b1n, ON, 100, 12800, 100);
      GEMM(ON, 100, encT[e][4], 104, bb1, nullptr, 0, F1, 50, Mtok, 100, 50, 1.f, 1);
      GEMM(F1, 50, encT[e][5], 56, bb2, ON, 100, F2, 100, Mtok, 50, 100, 1.f, 0);
      ln_kernel<<<Mtok, 128, 0, stream>>>(F2, g2, b2n, xout + (size_t)s0 * 100,
                                          100, 100, 25600);
    }
  };
  run_encoder(0, 0, xl_out);
  run_encoder(1, 100, xa_out);

  // -------------------- recurrent scan --------------------
  const float* wparam = IN(40);
  const float* S_b = IN(42);   const float* att_b = IN(44);
  const float* attl_b = IN(46); const float* atta_b = IN(48);
  const float* cm_b = IN(50);  const float* fcz_b = IN(52);
  const float* o1_b = IN(54);  const float* o2_b = IN(56);

  for (int t = 0; t < 128; ++t) {
    qprep_kernel<<<(256 * 512 + 255) / 256, 256, 0, stream>>>(qmask, qst, q0, t);
    GEMM(q0, 512, S_wT, 512, S_b, nullptr, 0, qaff, 512, 256, 512, 512, 1.f, 0);
    xcat_kernel<<<(2 * 256 * 484 + 255) / 256, 256, 0, stream>>>(xl_out, xa_out, allhs,
                                                                 XcatL, XcatA, t);
    GEMM(XcatL, 484, WUVlT, 488, bWUVl, qaff, 512, gatesL, 512, 256, 484, 512, 1.f, 0);
    GEMM(XcatA, 484, WUVaT, 488, bWUVa, qaff, 512, gatesA, 512, 256, 484, 512, 1.f, 0);
    qupdate_kernel<<<(256 * 1024 + 255) / 256, 256, 0, stream>>>(qmask, q0, qaff, qst, t);
    lsthm_kernel<<<(256 * 256 + 255) / 256, 256, 0, stream>>>(gatesL, gatesA, c_cat, allhs);
    GEMM(c_cat, 256, att_wT, 256, att_b, nullptr, 0, attraw, 1024, 256, 256, 1024, 1.f, 0);
    GEMM(c_cat, 256, attl_wT, 128, attl_b, nullptr, 0, attlraw, 512, 256, 128, 512, 1.f, 0);
    GEMM(c_cat + 128, 256, atta_wT, 128, atta_b, nullptr, 0, attaraw, 512, 256, 128, 512, 1.f, 0);
    attend_kernel<<<1024, 128, 0, stream>>>(attraw, attlraw, attaraw, c_cat, crossin, zflat);
    GEMM(crossin, 256, cm_wT, 256, cm_b, nullptr, 0, crossbuf, 144, 1024, 256, 144, 1.f, 1);
    zfill_kernel<<<(1024 * 144 + 255) / 256, 256, 0, stream>>>(crossbuf, zflat);
    GEMM(zflat, 1600, fcz_wT, 1600, fcz_b, nullptr, 0, zrelu, 256, 256, 1600, 256, 1.f, 1);
    znew_kernel<<<(256 * 256 + 255) / 256, 256, 0, stream>>>(wparam, zrelu, allhs);
    GEMM(allhs, 512, o1_wT, 512, o1_b, nullptr, 0, hid, 32, 256, 512, 32, 1.f, 1);
    GEMM(hid, 32, o2_wT, 32, o2_b, nullptr, 0, out + (size_t)t * 6, 768, 256, 32, 6, 1.f, 0);
  }
}
